// TransformerDiscriminator_86560770883817
// MI455X (gfx1250) — matrix operand order repaired and checked by reading, not compiled, not run on hardware
//
#include <hip/hip_runtime.h>
#include <hip/hip_bf16.h>

typedef _Float16 h16;
typedef __attribute__((ext_vector_type(16))) _Float16 v16h;
typedef __attribute__((ext_vector_type(8)))  float    v8f;

// Problem dims (compile-time)
#define BBATCH 4
#define TT 1024
#define DDIM 1024
#define HHEADS 8
#define LLAYERS 3
#define FFDIM 4096
#define HDH 128
#define D3 (3 * DDIM)

// GEMM tiling: 256 threads = 8 waves; block tile 64x128; wave tile 32x32 (2x2 WMMA)
#define BM 64
#define BN 128
#define BK 32
#define PADK 8

enum { EPI_SCORE = 0, EPI_H = 1, EPI_GELU = 2, EPI_RES = 3, EPI_POS = 4 };

// ---- CDNA5 async global->LDS copy (ASYNCcnt-tracked), with sync fallback ----
#if defined(__gfx1250__) && __has_builtin(__builtin_amdgcn_global_load_async_to_lds_b128)
#define HAS_ASYNC_LDS 1
typedef int v4i __attribute__((__vector_size__(16)));
typedef __attribute__((address_space(1))) v4i gv4i_t;
typedef __attribute__((address_space(3))) v4i lv4i_t;
#endif

__device__ __forceinline__ void async_copy16(h16* lds_dst, const h16* gsrc) {
#if defined(HAS_ASYNC_LDS)
    __builtin_amdgcn_global_load_async_to_lds_b128((gv4i_t*)gsrc, (lv4i_t*)lds_dst, 0, 0);
#else
    *reinterpret_cast<uint4*>(lds_dst) = *reinterpret_cast<const uint4*>(gsrc);
#endif
}

__device__ __forceinline__ void async_wait0() {
#if defined(HAS_ASYNC_LDS)
#if __has_builtin(__builtin_amdgcn_s_wait_asynccnt)
    __builtin_amdgcn_s_wait_asynccnt(0);
#else
    asm volatile("s_wait_asynccnt 0x0" ::: "memory");
#endif
#endif
}

struct GemmP {
    const h16* A;        // [M][K] row-major (lda)
    const h16* B;        // [N][K] row-major (ldb) -- k-contiguous rows
    float*     Cf;
    h16*       Ch;
    const float* bias;   // per output column, may be null
    const float* resid;  // residual, same layout/offsets as C (EPI_RES)
    int M, N, K, lda, ldb, ldc, Hz;
    long long sAb, sAh, sBb, sBh, sCb, sCh;  // batch strides: z = zb*Hz + zh
    float scale;
};

template <int EPI>
__global__ __launch_bounds__(256) void gemm_kernel(GemmP p) {
    // Double-buffered LDS; both operands stored k-contiguous: A[2][BM][.], B[2][BN][.]
    __shared__ __align__(16) h16 smem[2 * BM * (BK + PADK) + 2 * BN * (BK + PADK)];
    h16 (*As)[BK + PADK] = reinterpret_cast<h16(*)[BK + PADK]>(smem);
    h16 (*Bs)[BK + PADK] = reinterpret_cast<h16(*)[BK + PADK]>(smem + 2 * BM * (BK + PADK));

    const int tid  = threadIdx.x;
    const int lane = tid & 31;
    const int wave = tid >> 5;
    const int wm   = wave >> 2;   // 0..1
    const int wn   = wave & 3;    // 0..3
    const int hfl  = lane >> 4;   // lane half
    const int l16  = lane & 15;

    const int zb = (int)blockIdx.z / p.Hz;
    const int zh = (int)blockIdx.z % p.Hz;
    const int m0 = blockIdx.y * BM;
    const int n0 = blockIdx.x * BN;

    const h16* Ab = p.A + zb * p.sAb + zh * p.sAh + (long long)m0 * p.lda;
    const h16* Bb = p.B + zb * p.sBb + zh * p.sBh + (long long)n0 * p.ldb;
    const long long cb = zb * p.sCb + zh * p.sCh;

    // per-thread staging coordinates
    const int ar = tid >> 2, ac = (tid & 3) * 8;     // A: 64 rows x 4x8 halfs (1x16B)
    const int br = tid >> 1, bc = (tid & 1) * 16;    // B: 128 rows x 2x16 halfs (2x16B)

    auto stage = [&](int buf, int k0) {
        async_copy16(&As[buf * BM + ar][ac], Ab + (long long)ar * p.lda + k0 + ac);
        const h16* s = Bb + (long long)br * p.ldb + k0 + bc;
        async_copy16(&Bs[buf * BN + br][bc], s);
        async_copy16(&Bs[buf * BN + br][bc + 8], s + 8);
    };

    v8f acc[2][2] = {};

    const int nk = p.K / BK;
    stage(0, 0);
    for (int kt = 0; kt < nk; kt++) {
        const int buf = kt & 1;
        async_wait0();       // this wave's staged copies complete
        __syncthreads();     // all waves' copies visible; all waves done computing buf^1
        if (kt + 1 < nk) stage(buf ^ 1, (kt + 1) * BK);

        // --- fragments (ISA 16-bit layouts); 2x ds_load_b128 each ---
        v16h af[2], bf[2];
        const int kb = hfl ? 8 : 0;  // A: lanes 0-15 -> k 0..7/16..23, lanes 16-31 -> 8..15/24..31
#pragma unroll
        for (int f = 0; f < 2; f++) {
            const int row = buf * BM + wm * 32 + f * 16 + l16;
            union { uint4 u[2]; v16h v; } t;
            t.u[0] = *reinterpret_cast<const uint4*>(&As[row][kb]);
            t.u[1] = *reinterpret_cast<const uint4*>(&As[row][kb + 16]);
            af[f] = t.v;
        }
#pragma unroll
        for (int f = 0; f < 2; f++) {
            const int col = buf * BN + wn * 32 + f * 16 + l16;  // B: lanes 0-15 -> k 0..7/16..23, lanes 16-31 -> 8..15/24..31
            union { uint4 u[2]; v16h v; } t;
            t.u[0] = *reinterpret_cast<const uint4*>(&Bs[col][hfl * 8]);
            t.u[1] = *reinterpret_cast<const uint4*>(&Bs[col][hfl * 8 + 16]);
            bf[f] = t.v;
        }

#pragma unroll
        for (int i = 0; i < 2; i++)
#pragma unroll
            for (int j = 0; j < 2; j++)
                acc[i][j] = __builtin_amdgcn_wmma_f32_16x16x32_f16(
                    false, af[i], false, bf[j], (short)0, acc[i][j], false, false);
    }

    // --- fused epilogue; C layout: lanes 0-15 VGPR r -> M=r, lanes 16-31 -> M=r+8 ---
#pragma unroll
    for (int i = 0; i < 2; i++)
#pragma unroll
        for (int j = 0; j < 2; j++) {
            const int nn = n0 + wn * 32 + j * 16 + l16;
            const float bs = (EPI == EPI_SCORE) ? 0.f : (p.bias ? p.bias[nn] : 0.f);
#pragma unroll
            for (int r = 0; r < 8; r++) {
                const int m = m0 + wm * 32 + i * 16 + hfl * 8 + r;
                float v = acc[i][j][r];
                const long long idx = cb + (long long)m * p.ldc + nn;
                if (EPI == EPI_SCORE) {
                    p.Cf[idx] = v * p.scale;
                } else if (EPI == EPI_H) {
                    p.Ch[idx] = (h16)(v + bs);
                } else if (EPI == EPI_GELU) {
                    float x = v + bs;
                    p.Ch[idx] = (h16)(0.5f * x * (1.f + erff(x * 0.70710678118f)));
                } else if (EPI == EPI_RES) {
                    p.Cf[idx] = v + bs + p.resid[idx];
                } else {  // EPI_POS: sinusoidal positional encoding
                    const int   t   = m & (TT - 1);
                    const int   i2  = nn & ~1;
                    const float ang = (float)t *
                        __expf(-((float)i2 / (float)DDIM) * 9.210340371976184f);  // ln(1e4)
                    const float pe = (nn & 1) ? __cosf(ang) : __sinf(ang);
                    const float x  = v + bs + pe;
                    p.Cf[idx] = x;
                    p.Ch[idx] = (h16)x;
                }
            }
        }
}

// f32 -> f16 with transpose: src [z][K][N] row-major -> dst [z][N][K] row-major.
__global__ __launch_bounds__(256) void cvt_t_kernel(const float* src, h16* dst, int K, int N) {
    __shared__ h16 tile[32][33];
    const long long z = blockIdx.z;
    const float* s = src + z * (long long)K * N;
    h16* d = dst + z * (long long)K * N;
    const int k0 = blockIdx.x * 32, n0 = blockIdx.y * 32;
    const int tx = threadIdx.x & 31, ty = threadIdx.x >> 5;
#pragma unroll
    for (int i = 0; i < 4; i++) {
        const int k = k0 + ty + i * 8;
        tile[ty + i * 8][tx] = (h16)s[(long long)k * N + n0 + tx];
    }
    __syncthreads();
#pragma unroll
    for (int i = 0; i < 4; i++) {
        const int n = n0 + ty + i * 8;
        d[(long long)n * K + k0 + tx] = tile[tx][ty + i * 8];
    }
}

// V slice of qkv -> vt[(b*H+h)][hd][t]  (k-contiguous rows for aw@V)
__global__ __launch_bounds__(256) void transpose_v_kernel(const h16* qkv, h16* vt) {
    __shared__ h16 tile[32][33];
    const int z = blockIdx.z;                 // b*H + h
    const int b = z / HHEADS, h = z % HHEADS;
    const int t0 = blockIdx.x * 32, hd0 = blockIdx.y * 32;
    const int tx = threadIdx.x & 31, ty = threadIdx.x >> 5;
    const h16* src = qkv + (long long)b * TT * D3 + 2 * DDIM + h * HDH;
#pragma unroll
    for (int i = 0; i < 4; i++) {
        const int t = t0 + ty + i * 8;
        tile[ty + i * 8][tx] = src[(long long)t * D3 + hd0 + tx];
    }
    __syncthreads();
    h16* dst = vt + (long long)z * HDH * TT;
#pragma unroll
    for (int i = 0; i < 4; i++) {
        const int hd = hd0 + ty + i * 8;
        dst[(long long)hd * TT + t0 + tx] = tile[tx][ty + i * 8];
    }
}

// Softmax over key dim + 5-tap Gaussian smoothing, one block per (b,h,t) row
__global__ __launch_bounds__(256) void softmax_smooth_kernel(const float* scores, h16* aw) {
    __shared__ float pr[TT];
    __shared__ float red[256];
    const long long row = blockIdx.x;
    const float* src = scores + row * (long long)TT;
    const int tid = threadIdx.x;

    float v[4];
    float mx = -1e30f;
#pragma unroll
    for (int i = 0; i < 4; i++) { v[i] = src[tid + i * 256]; mx = fmaxf(mx, v[i]); }
    red[tid] = mx; __syncthreads();
    for (int s = 128; s > 0; s >>= 1) {
        if (tid < s) red[tid] = fmaxf(red[tid], red[tid + s]);
        __syncthreads();
    }
    mx = red[0]; __syncthreads();

    float sum = 0.f;
#pragma unroll
    for (int i = 0; i < 4; i++) { v[i] = __expf(v[i] - mx); sum += v[i]; }
    red[tid] = sum; __syncthreads();
    for (int s = 128; s > 0; s >>= 1) {
        if (tid < s) red[tid] += red[tid + s];
        __syncthreads();
    }
    const float inv = 1.f / red[0];
#pragma unroll
    for (int i = 0; i < 4; i++) pr[tid + i * 256] = v[i] * inv;
    __syncthreads();

    float w[5], ws = 0.f;
#pragma unroll
    for (int i = 0; i < 5; i++) { float d = (float)(i - 2); w[i] = __expf(-0.5f * d * d); ws += w[i]; }
#pragma unroll
    for (int i = 0; i < 5; i++) w[i] /= ws;

    h16* dst = aw + row * (long long)TT;
#pragma unroll
    for (int i = 0; i < 4; i++) {
        const int s = tid + i * 256;
        float o = 0.f;
#pragma unroll
        for (int k = 0; k < 5; k++) {
            const int q = s + k - 2;
            if (q >= 0 && q < TT) o += w[k] * pr[q];
        }
        dst[s] = (h16)o;
    }
}

// LayerNorm over D; writes fp32 master + f16 mirror. One block per row.
__global__ __launch_bounds__(256) void layernorm_kernel(const float* in, const float* g,
                                                        const float* b, float* xf, h16* xh) {
    __shared__ float red[256];
    const long long row = blockIdx.x;
    const float* src = in + row * (long long)DDIM;
    const int tid = threadIdx.x;

    float v[4];
    float s = 0.f;
#pragma unroll
    for (int i = 0; i < 4; i++) { v[i] = src[tid + i * 256]; s += v[i]; }
    red[tid] = s; __syncthreads();
    for (int st = 128; st > 0; st >>= 1) {
        if (tid < st) red[tid] += red[tid + st];
        __syncthreads();
    }
    const float mu = red[0] * (1.f / (float)DDIM);
    __syncthreads();

    float s2 = 0.f;
#pragma unroll
    for (int i = 0; i < 4; i++) { float d = v[i] - mu; s2 += d * d; }
    red[tid] = s2; __syncthreads();
    for (int st = 128; st > 0; st >>= 1) {
        if (tid < st) red[tid] += red[tid + st];
        __syncthreads();
    }
    const float rs = rsqrtf(red[0] * (1.f / (float)DDIM) + 1e-5f);
#pragma unroll
    for (int i = 0; i < 4; i++) {
        const int d = tid + i * 256;
        const float o = (v[i] - mu) * rs * g[d] + b[d];
        xf[row * (long long)DDIM + d] = o;
        xh[row * (long long)DDIM + d] = (h16)o;
    }
}

__global__ void meanpool_kernel(const float* xf, float* pooled) {
    const int d = blockIdx.x * blockDim.x + threadIdx.x;
    const int b = blockIdx.y;
    float s = 0.f;
    for (int t = 0; t < TT; t++) s += xf[((long long)(b * TT + t)) * DDIM + d];
    pooled[b * DDIM + d] = s * (1.f / (float)TT);
}

__global__ __launch_bounds__(256) void classifier_kernel(const float* pooled, const float* w1,
                                                         const float* b1, const float* w2,
                                                         const float* b2, float* out) {
    __shared__ float pl[DDIM];
    __shared__ float hid[DDIM / 2];
    __shared__ float r0[256], r1[256];
    const int b = blockIdx.x, tid = threadIdx.x;
    for (int i = tid; i < DDIM; i += 256) pl[i] = pooled[b * DDIM + i];
    __syncthreads();
    for (int j = tid; j < DDIM / 2; j += 256) {
        float s = b1[j];
        for (int i = 0; i < DDIM; i++) s += pl[i] * w1[i * (DDIM / 2) + j];
        hid[j] = fmaxf(s, 0.f);
    }
    __syncthreads();
    float s0 = 0.f, s1 = 0.f;
    for (int j = tid; j < DDIM / 2; j += 256) { s0 += hid[j] * w2[j * 2 + 0]; s1 += hid[j] * w2[j * 2 + 1]; }
    r0[tid] = s0; r1[tid] = s1; __syncthreads();
    for (int st = 128; st > 0; st >>= 1) {
        if (tid < st) { r0[tid] += r0[tid + st]; r1[tid] += r1[tid + st]; }
        __syncthreads();
    }
    if (tid == 0) { out[b * 2 + 0] = r0[0] + b2[0]; out[b * 2 + 1] = r1[0] + b2[1]; }
}

__global__ void cvt_kernel(const float* src, h16* dst, long long n) {
    const long long i = (long long)blockIdx.x * blockDim.x + threadIdx.x;
    if (i < n) dst[i] = (h16)src[i];
}

extern "C" void kernel_launch(void* const* d_in, const int* in_sizes, int n_in,
                              void* d_out, int out_size, void* d_ws, size_t ws_size,
                              hipStream_t stream) {
    (void)in_sizes; (void)n_in; (void)out_size; (void)ws_size;
    const float* x     = (const float*)d_in[0];
    const float* w_emb = (const float*)d_in[1];
    const float* b_emb = (const float*)d_in[2];
    const float* w_qkv = (const float*)d_in[3];
    const float* b_qkv = (const float*)d_in[4];
    const float* w_out = (const float*)d_in[5];
    const float* b_out = (const float*)d_in[6];
    const float* ln1g  = (const float*)d_in[7];
    const float* ln1b  = (const float*)d_in[8];
    const float* w_ff1 = (const float*)d_in[9];
    const float* b_ff1 = (const float*)d_in[10];
    const float* w_ff2 = (const float*)d_in[11];
    const float* b_ff2 = (const float*)d_in[12];
    const float* ln2g  = (const float*)d_in[13];
    const float* ln2b  = (const float*)d_in[14];
    const float* w_c1  = (const float*)d_in[15];
    const float* b_c1  = (const float*)d_in[16];
    const float* w_c2  = (const float*)d_in[17];
    const float* b_c2  = (const float*)d_in[18];

    char* ws = (char*)d_ws;
    size_t off = 0;
    auto alloc = [&](size_t bytes) -> char* {
        char* p = ws + off;
        off = (off + bytes + 255) & ~(size_t)255;
        return p;
    };

    const long long R = (long long)BBATCH * TT;  // 4096 rows
    h16*   xin    = (h16*)alloc(R * DDIM * 2);
    h16*   xh     = (h16*)alloc(R * DDIM * 2);
    float* xf     = (float*)alloc(R * DDIM * 4);
    float* tmp    = (float*)alloc(R * DDIM * 4);
    h16*   qkvh   = (h16*)alloc(R * D3 * 2);
    h16*   vth    = (h16*)alloc(R * DDIM * 2);     // v transposed: [(b*H+h)][hd][t]
    float* sc     = (float*)alloc((long long)BBATCH * HHEADS * TT * TT * 4);
    h16*   awh    = (h16*)alloc((long long)BBATCH * HHEADS * TT * TT * 2);
    h16*   atth   = (h16*)alloc(R * DDIM * 2);
    h16*   ffh    = (h16*)alloc(R * FFDIM * 2);
    float* pooled = (float*)alloc(BBATCH * DDIM * 4);
    h16*   wembt  = (h16*)alloc((long long)DDIM * DDIM * 2);
    h16*   wqkvt  = (h16*)alloc((long long)LLAYERS * DDIM * D3 * 2);
    h16*   woutt  = (h16*)alloc((long long)LLAYERS * DDIM * DDIM * 2);
    h16*   wff1t  = (h16*)alloc((long long)LLAYERS * DDIM * FFDIM * 2);
    h16*   wff2t  = (h16*)alloc((long long)LLAYERS * FFDIM * DDIM * 2);

    // Convert activations (plain) and weights (f32->f16 + transpose to [N][K])
    cvt_kernel<<<dim3((unsigned)((R * DDIM + 255) / 256)), 256, 0, stream>>>(x, xin, R * DDIM);
    cvt_t_kernel<<<dim3(DDIM / 32, DDIM / 32, 1), 256, 0, stream>>>(w_emb, wembt, DDIM, DDIM);
    cvt_t_kernel<<<dim3(DDIM / 32, D3 / 32, LLAYERS), 256, 0, stream>>>(w_qkv, wqkvt, DDIM, D3);
    cvt_t_kernel<<<dim3(DDIM / 32, DDIM / 32, LLAYERS), 256, 0, stream>>>(w_out, woutt, DDIM, DDIM);
    cvt_t_kernel<<<dim3(DDIM / 32, FFDIM / 32, LLAYERS), 256, 0, stream>>>(w_ff1, wff1t, DDIM, FFDIM);
    cvt_t_kernel<<<dim3(FFDIM / 32, DDIM / 32, LLAYERS), 256, 0, stream>>>(w_ff2, wff2t, FFDIM, DDIM);

    // Embedding: x @ w_emb + b_emb + pos  -> xf (fp32 master) + xh (f16 mirror)
    {
        GemmP p{};
        p.A = xin; p.B = wembt; p.Cf = xf; p.Ch = xh; p.bias = b_emb;
        p.M = (int)R; p.N = DDIM; p.K = DDIM;
        p.lda = DDIM; p.ldb = DDIM; p.ldc = DDIM; p.Hz = 1; p.scale = 1.f;
        gemm_kernel<EPI_POS><<<dim3(DDIM / BN, (unsigned)(R / BM), 1), 256, 0, stream>>>(p);
    }

    for (int l = 0; l < LLAYERS; l++) {
        // qkv = xh @ w_qkv[l] + b_qkv[l]  (f16 out)
        {
            GemmP p{};
            p.A = xh; p.B = wqkvt + (long long)l * DDIM * D3; p.Ch = qkvh;
            p.bias = b_qkv + l * D3;
            p.M = (int)R; p.N = D3; p.K = DDIM;
            p.lda = DDIM; p.ldb = DDIM; p.ldc = D3; p.Hz = 1; p.scale = 1.f;
            gemm_kernel<EPI_H><<<dim3(D3 / BN, (unsigned)(R / BM), 1), 256, 0, stream>>>(p);
        }
        // v slice -> vt[(b*H+h)][hd][t]
        transpose_v_kernel<<<dim3(TT / 32, HDH / 32, BBATCH * HHEADS), 256, 0, stream>>>(qkvh, vth);
        // scores = q @ k^T * scale, batched over (b,h); K-matrix rows are k-contiguous
        {
            GemmP p{};
            p.A = qkvh; p.B = qkvh + DDIM; p.Cf = sc;
            p.M = TT; p.N = TT; p.K = HDH;
            p.lda = D3; p.ldb = D3; p.ldc = TT; p.Hz = HHEADS;
            p.sAb = (long long)TT * D3; p.sAh = HDH;
            p.sBb = (long long)TT * D3; p.sBh = HDH;
            p.sCb = (long long)HHEADS * TT * TT; p.sCh = (long long)TT * TT;
            p.scale = 0.08838834764831845f;  // 1/sqrt(128)
            gemm_kernel<EPI_SCORE><<<dim3(TT / BN, TT / BM, BBATCH * HHEADS), 256, 0, stream>>>(p);
        }
        // softmax over keys + Gaussian smoothing -> aw (f16)
        softmax_smooth_kernel<<<dim3(BBATCH * HHEADS * TT), 256, 0, stream>>>(sc, awh);
        // attn = aw @ v, batched over (b,h); B = vt (k-contiguous); out (b,t,h*HD+:) f16
        {
            GemmP p{};
            p.A = awh; p.B = vth; p.Ch = atth;
            p.M = TT; p.N = HDH; p.K = TT;
            p.lda = TT; p.ldb = TT; p.ldc = DDIM; p.Hz = HHEADS;
            p.sAb = (long long)HHEADS * TT * TT; p.sAh = (long long)TT * TT;
            p.sBb = (long long)HHEADS * HDH * TT; p.sBh = (long long)HDH * TT;
            p.sCb = (long long)TT * DDIM; p.sCh = HDH;
            p.scale = 1.f;
            gemm_kernel<EPI_H><<<dim3(HDH / BN, TT / BM, BBATCH * HHEADS), 256, 0, stream>>>(p);
        }
        // x + attn @ w_out + b_out -> tmp (pre-LN fp32)
        {
            GemmP p{};
            p.A = atth; p.B = woutt + (long long)l * DDIM * DDIM; p.Cf = tmp;
            p.bias = b_out + l * DDIM; p.resid = xf;
            p.M = (int)R; p.N = DDIM; p.K = DDIM;
            p.lda = DDIM; p.ldb = DDIM; p.ldc = DDIM; p.Hz = 1; p.scale = 1.f;
            gemm_kernel<EPI_RES><<<dim3(DDIM / BN, (unsigned)(R / BM), 1), 256, 0, stream>>>(p);
        }
        layernorm_kernel<<<dim3((unsigned)R), 256, 0, stream>>>(
            tmp, ln1g + l * DDIM, ln1b + l * DDIM, xf, xh);
        // ff1: gelu(xh @ w_ff1 + b_ff1) -> ffh (f16)
        {
            GemmP p{};
            p.A = xh; p.B = wff1t + (long long)l * DDIM * FFDIM; p.Ch = ffh;
            p.bias = b_ff1 + l * FFDIM;
            p.M = (int)R; p.N = FFDIM; p.K = DDIM;
            p.lda = DDIM; p.ldb = DDIM; p.ldc = FFDIM; p.Hz = 1; p.scale = 1.f;
            gemm_kernel<EPI_GELU><<<dim3(FFDIM / BN, (unsigned)(R / BM), 1), 256, 0, stream>>>(p);
        }
        // ff2 + residual -> tmp
        {
            GemmP p{};
            p.A = ffh; p.B = wff2t + (long long)l * FFDIM * DDIM; p.Cf = tmp;
            p.bias = b_ff2 + l * DDIM; p.resid = xf;
            p.M = (int)R; p.N = DDIM; p.K = FFDIM;
            p.lda = FFDIM; p.ldb = FFDIM; p.ldc = DDIM; p.Hz = 1; p.scale = 1.f;
            gemm_kernel<EPI_RES><<<dim3(DDIM / BN, (unsigned)(R / BM), 1), 256, 0, stream>>>(p);
        }
        layernorm_kernel<<<dim3((unsigned)R), 256, 0, stream>>>(
            tmp, ln2g + l * DDIM, ln2b + l * DDIM, xf, xh);
    }

    meanpool_kernel<<<dim3(DDIM / 256, BBATCH), 256, 0, stream>>>(xf, pooled);
    classifier_kernel<<<dim3(BBATCH), 256, 0, stream>>>(pooled, w_c1, b_c1, w_c2, b_c2,
                                                        (float*)d_out);
}
